// Attenion_35381940584533
// MI455X (gfx1250) — matrix operand order repaired and checked by reading, not compiled, not run on hardware
//
#include <hip/hip_runtime.h>
#include <hip/hip_bf16.h>
#include <math.h>

typedef __attribute__((ext_vector_type(16))) _Float16 v16h;
typedef __attribute__((ext_vector_type(8)))  float    v8f;

#define B_ 256
#define S_ 512
#define D_ 50
#define E_ 64
#define G_ 256   // 4*E gates

// ---------------------------------------------------------------- helpers ---

__device__ __forceinline__ v8f wmma_f16(v16h a, v16h b, v8f c) {
  // D = A(16x32 f16) * B(32x16 f16) + C(16x16 f32)
  return __builtin_amdgcn_wmma_f32_16x16x32_f16(false, a, false, b, (short)0, c,
                                                false, false);
}

// fast sigmoid/tanh: v_exp_f32 + v_rcp_f32 (avoids IEEE div_scale sequences)
__device__ __forceinline__ float sig_f(float x) {
  return __builtin_amdgcn_rcpf(1.0f + __expf(-x));
}
__device__ __forceinline__ float tanh_f(float x) { return 2.0f * sig_f(2.0f * x) - 1.0f; }

// A fragment (16x32 f16) from an LDS tile laid out [16 rows][64 cols] f16.
// This code's order (same as frag_b_T): lane holds row m = lane&15; half hh = lane>>4 selects a 16-wide k run:
// element e -> k = e + 16*hh (+32*kc). Two b128 LDS loads.
__device__ __forceinline__ v16h frag_a_lds(const _Float16* sm, int lane, int kc) {
  const int m = lane & 15, hh = lane >> 4;
  const _Float16* p = sm + m * 64 + kc * 32 + 16 * hh;
  union { uint4 u; _Float16 h[8]; } t0, t1;
  t0.u = *(const uint4*)(p);
  t1.u = *(const uint4*)(p + 8);
  v16h a;
#pragma unroll
  for (int i = 0; i < 8; ++i) { a[i] = t0.h[i]; a[8 + i] = t1.h[i]; }
  return a;
}

// A fragment from global hseq[S][B][E] (f32 -> f16): row = time index srow for batch b.
__device__ __forceinline__ v16h frag_a_hseq(const float* hseq, int b, int srow,
                                            int lane, int kc) {
  const int hh = lane >> 4;
  const float* p = hseq + ((size_t)srow * B_ + b) * E_ + kc * 32 + 16 * hh;
  v16h a;
#pragma unroll
  for (int i = 0; i < 8; ++i) { a[i] = (_Float16)p[i]; a[8 + i] = (_Float16)p[8 + i]; }
  return a;
}

// B fragment (32x16 f16) for GEMM x @ W^T:  B[k][n] = W[ncol][k], W row-major [N][K].
// Per lane: n = lane&15, element e -> k = kc*32 + 16*(lane>>4) + e  (contiguous row run).
__device__ __forceinline__ v16h frag_b_T(const float* W, int ld, int kmax,
                                         int ncol, int lane, int kc) {
  const int hh = lane >> 4, k0 = kc * 32 + 16 * hh;
  const float* p = W + (size_t)ncol * ld;
  v16h bf;
#pragma unroll
  for (int e = 0; e < 16; ++e) {
    int k = k0 + e;
    bf[e] = (k < kmax) ? (_Float16)p[k] : (_Float16)0.0f;
  }
  return bf;
}

// B fragment for GEMM h @ W (no transpose): B[k][n] = W[k][ncol], W row-major [K][N].
__device__ __forceinline__ v16h frag_b_plain(const float* W, int ld, int ncol,
                                             int lane, int kc) {
  const int hh = lane >> 4, k0 = kc * 32 + 16 * hh;
  v16h bf;
#pragma unroll
  for (int e = 0; e < 16; ++e) bf[e] = (_Float16)W[(size_t)(k0 + e) * ld + ncol];
  return bf;
}

// ------------------------------------------------------------ LSTM kernel ---
// 16 blocks x 256 threads (8 waves). Block owns 16 batch rows for all 512 steps.
// Weights resident in VGPRs, h/c resident in LDS. The embedding gather for step
// s+1 is software-pipelined: issued into registers before the WMMAs of step s,
// committed to the alternate s_x buffer during the elementwise phase. Inline
// take_along_axis gathers via gidx. 2 barriers per step.
__global__ __launch_bounds__(256) void lstm_kernel(
    const int* __restrict__ tokens,            // [B,S]
    const float* __restrict__ emb,             // [V,D]
    const float* __restrict__ w_ih,            // [G,D]
    const float* __restrict__ w_hh,            // [G,E]
    const float* __restrict__ b_ih, const float* __restrict__ b_hh,  // [G]
    const float* __restrict__ h0, const float* __restrict__ c0,      // [B,E] or null
    float* __restrict__ hseq,                  // [S,B,E] or null
    const int* __restrict__ gidx,              // [B,E] or null
    float* __restrict__ gh, float* __restrict__ gc)  // [B,E] or null
{
  __shared__ __align__(16) _Float16 s_x[2][16 * 64];  // x_t tiles (double buffered)
  __shared__ __align__(16) _Float16 s_h[16 * 64];     // h_{t-1} as f16 (WMMA A src)
  __shared__ float s_c[16 * 64];                      // cell state
  __shared__ float s_g[16 * G_];                      // gate pre-activations
  __shared__ int   s_tok[2][16];                      // tokens (double buffered)

  const int tid  = threadIdx.x;
  const int lane = tid & 31, wave = tid >> 5;
  const int bbase = blockIdx.x * 16;
  const int dcol = tid & 63;                          // feature column of this thread

  // init h, c
#pragma unroll
  for (int rep = 0; rep < 4; ++rep) {
    int cell = tid + rep * 256;
    int bl = cell >> 6, e = cell & 63;
    float hv = h0 ? h0[(bbase + bl) * E_ + e] : 0.0f;
    float cv = c0 ? c0[(bbase + bl) * E_ + e] : 0.0f;
    s_h[cell] = (_Float16)hv;
    s_c[cell] = cv;
  }

  // weight fragments resident in registers: wave owns gate tiles nt0, nt0+1
  const int nt0 = wave * 2;
  v16h Bih[2][2], Bhh[2][2];
  float biasv[2];
#pragma unroll
  for (int nt = 0; nt < 2; ++nt) {
    int ng = (nt0 + nt) * 16 + (lane & 15);
    biasv[nt] = b_ih[ng] + b_hh[ng];
#pragma unroll
    for (int kc = 0; kc < 2; ++kc) {
      Bih[nt][kc] = frag_b_T(w_ih, D_, D_, ng, lane, kc);  // K padded 50 -> 64
      Bhh[nt][kc] = frag_b_T(w_hh, E_, E_, ng, lane, kc);
    }
  }

  // gather indices for inline take_along_axis
  int gi[4];
#pragma unroll
  for (int rep = 0; rep < 4; ++rep) {
    int cell = tid + rep * 256;
    int bl = cell >> 6, e = cell & 63;
    gi[rep] = gidx ? gidx[(bbase + bl) * E_ + e] : -1;
  }

  // prime the pipeline: x_0 into s_x[0], tokens for s=1 into s_tok[0]
  if (tid < 16) s_tok[1][tid] = tokens[(bbase + tid) * S_];
  __syncthreads();
#pragma unroll
  for (int rep = 0; rep < 4; ++rep) {
    int cell = tid + rep * 256;
    int bl = (tid >> 6) + rep * 4;
    float xv = (dcol < D_) ? emb[(size_t)s_tok[1][bl] * D_ + dcol] : 0.0f;
    s_x[0][cell] = (_Float16)xv;
  }
  if (tid < 16) s_tok[0][tid] = tokens[(bbase + tid) * S_ + 1];
  __syncthreads();

  int cur = 0;
  for (int s = 0; s < S_; ++s) {
    const int nxt = cur ^ 1;

    // issue prefetch for step s+1 (embeddings) and s+2 (tokens); these global
    // loads drain while the WMMAs below execute.
    float px[4];
    int tnext = 0;
    if (tid < 16 && s + 2 < S_) tnext = tokens[(bbase + tid) * S_ + (s + 2)];
#pragma unroll
    for (int rep = 0; rep < 4; ++rep) {
      int bl = (tid >> 6) + rep * 4;
      px[rep] = (dcol < D_ && s + 1 < S_)
                    ? emb[(size_t)s_tok[cur][bl] * D_ + dcol]
                    : 0.0f;
    }

    // gates = bias + x@W_ih^T + h@W_hh^T  (4 chained WMMAs per gate tile)
#pragma unroll
    for (int nt = 0; nt < 2; ++nt) {
      v8f acc;
#pragma unroll
      for (int r = 0; r < 8; ++r) acc[r] = biasv[nt];
      acc = wmma_f16(frag_a_lds(s_x[cur], lane, 0), Bih[nt][0], acc);
      acc = wmma_f16(frag_a_lds(s_x[cur], lane, 1), Bih[nt][1], acc);
      acc = wmma_f16(frag_a_lds(s_h, lane, 0), Bhh[nt][0], acc);
      acc = wmma_f16(frag_a_lds(s_h, lane, 1), Bhh[nt][1], acc);
      int n  = (nt0 + nt) * 16 + (lane & 15);
      int mb = 8 * (lane >> 4);
#pragma unroll
      for (int r = 0; r < 8; ++r) s_g[(mb + r) * G_ + n] = acc[r];
    }
    __syncthreads();

    // elementwise LSTM cell update (PyTorch gate order i,f,g,o)
#pragma unroll
    for (int rep = 0; rep < 4; ++rep) {
      int cell = tid + rep * 256;
      int bl = cell >> 6, e = cell & 63;
      float g_i = s_g[bl * G_ + e];
      float g_f = s_g[bl * G_ + 64 + e];
      float g_g = s_g[bl * G_ + 128 + e];
      float g_o = s_g[bl * G_ + 192 + e];
      float cn = sig_f(g_f) * s_c[cell] + sig_f(g_i) * tanh_f(g_g);
      float hn = sig_f(g_o) * tanh_f(cn);
      s_c[cell] = cn;
      s_h[cell] = (_Float16)hn;
      if (hseq) hseq[((size_t)s * B_ + bbase + bl) * E_ + e] = hn;
      if (gi[rep] == s) {
        if (gh) gh[(bbase + bl) * E_ + e] = hn;
        if (gc) gc[(bbase + bl) * E_ + e] = cn;
      }
    }

    // commit prefetched x_{s+1} / tokens_{s+2} into the alternate buffers
#pragma unroll
    for (int rep = 0; rep < 4; ++rep) {
      int cell = tid + rep * 256;
      s_x[nxt][cell] = (_Float16)px[rep];
    }
    if (tid < 16) s_tok[nxt][tid] = tnext;
    __syncthreads();
    cur = nxt;
  }
}

// --------------------------------------------------- u = out @ wh  [B,64] ---
__global__ __launch_bounds__(256) void uproj_kernel(const float* __restrict__ outg,
                                                    const float* __restrict__ wh,
                                                    float* __restrict__ u) {
  int id = blockIdx.x * 256 + threadIdx.x;   // 64 blocks -> B_*E_ threads
  int b = id >> 6, n = id & 63;
  float acc = 0.0f;
#pragma unroll
  for (int k = 0; k < E_; ++k) acc += outg[b * E_ + k] * wh[k * E_ + n];
  u[b * E_ + n] = acc;
}

// ----------------- wtm[b,s] = mask( wvec . tanh(h[b,s]@wy + u[b]) ) ----------
__global__ __launch_bounds__(256) void attn_wtm_kernel(
    const float* __restrict__ hseq, const float* __restrict__ wy,
    const float* __restrict__ u, const float* __restrict__ wvec,
    const float* __restrict__ s1s, float* __restrict__ wtm) {
  const int tid = threadIdx.x, lane = tid & 31, wave = tid >> 5;
  const int b = blockIdx.x >> 2;                       // 4 blocks per batch row
  const int sbase = (blockIdx.x & 3) * 128 + wave * 16;
  const int srow = sbase + (lane & 15);

  v16h a0 = frag_a_hseq(hseq, b, srow, lane, 0);
  v16h a1 = frag_a_hseq(hseq, b, srow, lane, 1);

  float red[8];
#pragma unroll
  for (int r = 0; r < 8; ++r) red[r] = 0.0f;

#pragma unroll
  for (int nt = 0; nt < 4; ++nt) {
    v8f acc;
#pragma unroll
    for (int r = 0; r < 8; ++r) acc[r] = 0.0f;
    int ncol = nt * 16 + (lane & 15);
    acc = wmma_f16(a0, frag_b_plain(wy, E_, ncol, lane, 0), acc);
    acc = wmma_f16(a1, frag_b_plain(wy, E_, ncol, lane, 1), acc);
    float uu = u[b * E_ + ncol];
    float wv = wvec[ncol];
#pragma unroll
    for (int r = 0; r < 8; ++r) red[r] += tanh_f(acc[r] + uu) * wv;
  }
  // reduce the 16 n-lanes of each half -> one score per output row
#pragma unroll
  for (int r = 0; r < 8; ++r) {
    float v = red[r];
    v += __shfl_xor(v, 1);
    v += __shfl_xor(v, 2);
    v += __shfl_xor(v, 4);
    v += __shfl_xor(v, 8);
    if ((lane & 15) == 0) {
      int sr = sbase + r + 8 * (lane >> 4);
      float m = s1s[b * S_ + sr];
      wtm[b * S_ + sr] = m * v - (1.0f - m) * 1e12f;
    }
  }
}

// ------------- softmax over s and r[b] = sum_s p_s * h[b,s,:]  --------------
__global__ __launch_bounds__(256) void softmax_r_kernel(
    const float* __restrict__ wtm, const float* __restrict__ hseq,
    float* __restrict__ rbuf) {
  __shared__ float sp[512];
  __shared__ float sred[256];
  const int tid = threadIdx.x, b = blockIdx.x;

  float a0 = wtm[b * S_ + tid], a1 = wtm[b * S_ + 256 + tid];
  sred[tid] = fmaxf(a0, a1);
  __syncthreads();
  for (int st = 128; st > 0; st >>= 1) {
    if (tid < st) sred[tid] = fmaxf(sred[tid], sred[tid + st]);
    __syncthreads();
  }
  float mx = sred[0];
  __syncthreads();

  float e0 = __expf(a0 - mx), e1 = __expf(a1 - mx);
  sp[tid] = e0; sp[tid + 256] = e1;
  sred[tid] = e0 + e1;
  __syncthreads();
  for (int st = 128; st > 0; st >>= 1) {
    if (tid < st) sred[tid] += sred[tid + st];
    __syncthreads();
  }
  float inv = __builtin_amdgcn_rcpf(sred[0]);
  __syncthreads();

  // r: 4 s-strides x 64 features
  int e = tid & 63, sg = tid >> 6;
  float acc = 0.0f;
  for (int s = sg; s < S_; s += 4)
    acc += sp[s] * hseq[((size_t)s * B_ + b) * E_ + e];
  sred[tid] = acc;
  __syncthreads();
  if (tid < 64)
    rbuf[b * E_ + tid] =
        (sred[tid] + sred[64 + tid] + sred[128 + tid] + sred[192 + tid]) * inv;
}

// ------------------------------ final MLP head ------------------------------
__global__ __launch_bounds__(256) void head_kernel(
    const float* __restrict__ rbuf, const float* __restrict__ outg,
    const float* __restrict__ wp, const float* __restrict__ wx,
    const float* __restrict__ l1w, const float* __restrict__ l1b,
    const float* __restrict__ lw, const float* __restrict__ lb,
    float* __restrict__ out) {
  __shared__ float s_r[16 * 64], s_o[16 * 64], s_t1[16 * 64], s_t2[16 * 128];
  const int tid = threadIdx.x, bbase = blockIdx.x * 16;

#pragma unroll
  for (int rep = 0; rep < 4; ++rep) {
    int cell = tid + rep * 256;
    int bl = cell >> 6, e = cell & 63;
    s_r[cell] = rbuf[(bbase + bl) * E_ + e];
    s_o[cell] = outg[(bbase + bl) * E_ + e];
  }
  __syncthreads();

#pragma unroll
  for (int rep = 0; rep < 4; ++rep) {           // t1 = tanh(r@wp + out@wx)
    int cell = tid + rep * 256;
    int bl = cell >> 6, j = cell & 63;
    float acc = 0.0f;
    for (int k = 0; k < E_; ++k)
      acc += s_r[bl * 64 + k] * wp[k * E_ + j] + s_o[bl * 64 + k] * wx[k * E_ + j];
    s_t1[cell] = tanh_f(acc);
  }
  __syncthreads();

#pragma unroll
  for (int rep = 0; rep < 8; ++rep) {           // t2 = tanh(t1 @ l1_w^T + l1_b)
    int cell = tid + rep * 256;
    int bl = cell >> 7, j = cell & 127;
    float acc = l1b[j];
    for (int k = 0; k < E_; ++k) acc += s_t1[bl * 64 + k] * l1w[j * E_ + k];
    s_t2[cell] = tanh_f(acc);
  }
  __syncthreads();

  if (tid < 64) {                               // logits = t2 @ l_w^T + l_b
    int bl = tid >> 2, q = tid & 3;
    float acc = lb[q];
    for (int k = 0; k < 128; ++k) acc += s_t2[bl * 128 + k] * lw[q * 128 + k];
    out[(bbase + bl) * 4 + q] = acc;
  }
}

// ---------------------------------------------------------------------------

extern "C" void kernel_launch(void* const* d_in, const int* in_sizes, int n_in,
                              void* d_out, int out_size, void* d_ws, size_t ws_size,
                              hipStream_t stream) {
  (void)in_sizes; (void)n_in; (void)out_size; (void)ws_size;
  const int*   sent1  = (const int*)d_in[0];
  const int*   sent2  = (const int*)d_in[1];
  const int*   s1_len = (const int*)d_in[2];
  const int*   s2_len = (const int*)d_in[3];
  const float* s1_s   = (const float*)d_in[4];
  // d_in[5] (s2_s) unused by the reference math path
  const float* emb    = (const float*)d_in[6];
  const float* w_ih1  = (const float*)d_in[7];
  const float* w_hh1  = (const float*)d_in[8];
  const float* b_ih1  = (const float*)d_in[9];
  const float* b_hh1  = (const float*)d_in[10];
  const float* w_ih2  = (const float*)d_in[11];
  const float* w_hh2  = (const float*)d_in[12];
  const float* b_ih2  = (const float*)d_in[13];
  const float* b_hh2  = (const float*)d_in[14];
  const float* wy     = (const float*)d_in[15];
  const float* wh     = (const float*)d_in[16];
  const float* wvec   = (const float*)d_in[17];
  const float* wp     = (const float*)d_in[18];
  const float* wx     = (const float*)d_in[19];
  const float* l1_w   = (const float*)d_in[20];
  const float* l1_b   = (const float*)d_in[21];
  const float* l_w    = (const float*)d_in[22];
  const float* l_b    = (const float*)d_in[23];

  char* ws = (char*)d_ws;
  float* h1seq = (float*)ws; ws += (size_t)S_ * B_ * E_ * sizeof(float);  // 33.5MB
  float* gh    = (float*)ws; ws += B_ * E_ * sizeof(float);
  float* gc    = (float*)ws; ws += B_ * E_ * sizeof(float);
  float* outg  = (float*)ws; ws += B_ * E_ * sizeof(float);
  float* u     = (float*)ws; ws += B_ * E_ * sizeof(float);
  float* wtm   = (float*)ws; ws += (size_t)B_ * S_ * sizeof(float);
  float* rbuf  = (float*)ws; ws += B_ * E_ * sizeof(float);

  // LSTM1 over sentence1: zero init, store full h sequence, inline-gather h/c at s1_len
  lstm_kernel<<<16, 256, 0, stream>>>(sent1, emb, w_ih1, w_hh1, b_ih1, b_hh1,
                                      nullptr, nullptr, h1seq, s1_len, gh, gc);
  // LSTM2 over sentence2: init from gathered state, inline-gather h at s2_len -> outg
  lstm_kernel<<<16, 256, 0, stream>>>(sent2, emb, w_ih2, w_hh2, b_ih2, b_hh2,
                                      gh, gc, nullptr, s2_len, outg, nullptr);
  uproj_kernel<<<64, 256, 0, stream>>>(outg, wh, u);
  attn_wtm_kernel<<<1024, 256, 0, stream>>>(h1seq, wy, u, wvec, s1_s, wtm);
  softmax_r_kernel<<<256, 256, 0, stream>>>(wtm, h1seq, rbuf);
  head_kernel<<<16, 256, 0, stream>>>(rbuf, outg, wp, wx, l1_w, l1_b, l_w, l_b,
                                      (float*)d_out);
}
